// GAT0tampo_2302102470995
// MI455X (gfx1250) — matrix operand order repaired and checked by reading, not compiled, not run on hardware
//
#include <hip/hip_runtime.h>
#include <hip/hip_bf16.h>

// ---------------- problem constants (match reference) ----------------
#define NNODES 30000
#define NEDGES 150000
#define NGRAPH 64
#define NHEADS 4
#define HC1 1024   // H*C1
#define CC1 256
#define HC2 256    // H*C2
#define CC2 64
#define EDIM 16
#define KMAX 1024
#define KPAD 8     // LDS row pad (bf16 elems) to avoid bank conflicts

typedef __attribute__((ext_vector_type(8)))  __bf16 v8bf;
typedef __attribute__((ext_vector_type(16))) __bf16 v16bf;
typedef __attribute__((ext_vector_type(8)))  float  v8f;
typedef __attribute__((ext_vector_type(4)))  float  v4f;    // plain vector (not HIP class)
typedef int v4i_payload __attribute__((vector_size(16)));   // b128 payload type

#if __has_builtin(__builtin_amdgcn_global_load_async_to_lds_b128)
#define HAVE_ASYNC_BUILTIN 1
#else
#define HAVE_ASYNC_BUILTIN 0
#endif

// Load through an explicit global (address_space(1)) pointer so the backend
// emits global_load_* (LOADcnt only) instead of flat_load_* (LOADcnt+DScnt).
template <typename T>
__device__ inline T gload(const T* p) {
  return *(const __attribute__((address_space(1))) T*)p;
}

// global_prefetch_b8 on the global path (syntax matches gfx1250 disasm)
__device__ inline void gprefetch(const void* p) {
  asm volatile("global_prefetch_b8 %0, off scope:SCOPE_SE" :: "v"(p));
}

// 16-byte global -> LDS async copy (ASYNCcnt-tracked DMA, no VGPR round trip)
__device__ inline void cp16_g2l(__bf16* lds_dst, const __bf16* g_src) {
#if HAVE_ASYNC_BUILTIN
  __builtin_amdgcn_global_load_async_to_lds_b128(
      (__attribute__((address_space(1))) v4i_payload*)g_src,
      (__attribute__((address_space(3))) v4i_payload*)lds_dst,
      0, 0);
#else
  unsigned int loff =
      (unsigned int)(unsigned long long)(__attribute__((address_space(3))) void*)lds_dst;
  asm volatile("global_load_async_to_lds_b128 %0, %1, off"
               :: "v"(loff), "v"(g_src) : "memory");
#endif
}

__device__ inline void wait_async0() {
  asm volatile("s_wait_asynccnt 0x0" ::: "memory");
}

// ---------------- small helpers ----------------
__global__ void k_fill(float* __restrict__ p, float v, long n) {
  long i = (long)blockIdx.x * blockDim.x + threadIdx.x;
  if (i < n) p[i] = v;
}

__global__ void k_cast_bf16(const float* __restrict__ in, __bf16* __restrict__ out, long n) {
  long i = (long)blockIdx.x * blockDim.x + threadIdx.x;
  if (i < n) out[i] = (__bf16)in[i];
}

// Wt[n*K + k] = (bf16)W[k*Nout + n]   (pre-transpose so WMMA B frags are contiguous)
__global__ void k_transpose_cast(const float* __restrict__ W, __bf16* __restrict__ Wt,
                                 int K, int Nout) {
  long i = (long)blockIdx.x * blockDim.x + threadIdx.x;
  if (i >= (long)K * Nout) return;
  int k = (int)(i / Nout), n = (int)(i % Nout);
  Wt[(size_t)n * K + k] = (__bf16)W[i];
}

// Wt[n*K + k] = W[k*Nout + n]  (fp32; makes We rows contiguous per channel)
__global__ void k_transpose_f32(const float* __restrict__ W, float* __restrict__ Wt,
                                int K, int Nout) {
  long i = (long)blockIdx.x * blockDim.x + threadIdx.x;
  if (i >= (long)K * Nout) return;
  int k = (int)(i / Nout), n = (int)(i % Nout);
  Wt[(size_t)n * K + k] = W[i];
}

__global__ void k_bias_init(const float* __restrict__ bias, float* __restrict__ out,
                            int HC, long n) {
  long i = (long)blockIdx.x * blockDim.x + threadIdx.x;
  if (i < n) out[i] = bias[i % HC];
}

__global__ void k_relu_cast(float* __restrict__ x, __bf16* __restrict__ xb, long n) {
  long i = (long)blockIdx.x * blockDim.x + threadIdx.x;
  if (i < n) {
    float v = x[i];
    v = v > 0.0f ? v : 0.0f;
    x[i] = v;
    if (xb) xb[i] = (__bf16)v;
  }
}

__device__ inline void atomicMaxF(float* addr, float val) {
  int* ai = (int*)addr;
  int cur = *ai;
  while (__int_as_float(cur) < val) {
    int prev = atomicCAS(ai, cur, __float_as_int(val));
    if (prev == cur) break;
    cur = prev;
  }
}

// ---------------- WMMA GEMM: C[M,Nout] = A_bf16[M,K] @ B + bias ----------------
// Bt[Nout,K] bf16 (pre-transposed). Block = 256 threads = 8 waves.
// A tile (16 rows x K) staged once per block in LDS via async DMA and shared by
// all 8 waves. Each wave register-blocks NT consecutive 16x16 N-tiles: one A
// fragment (from LDS) feeds NT WMMAs. Block covers 16 rows x (8*16*NT) cols.
template <int NT>
__global__ void __launch_bounds__(256)
k_gemm_wmma_bf16(const __bf16* __restrict__ A, const __bf16* __restrict__ Bt,
                 const float* __restrict__ bias, float* __restrict__ C,
                 int M, int K, int Nout) {
  __shared__ __bf16 sA[16 * (KMAX + KPAD)];
  const int tid  = threadIdx.x;
  const int lane = tid & 31;
  const int wave = tid >> 5;
  const int tm   = blockIdx.x << 4;
  const int tn   = ((blockIdx.y << 3) + wave) * (16 * NT);
  const int l15  = lane & 15;
  const int hi   = lane >> 4;
  const int KP   = K + KPAD;

  // ---- stage A tile [16 x K] into LDS with async global->LDS DMA ----
  const int kc = K >> 3;                 // 16B chunks per row
  for (int idx = tid; idx < 16 * kc; idx += 256) {
    int r = idx / kc, c8 = (idx % kc) << 3;
    cp16_g2l(&sA[r * KP + c8], A + (size_t)(tm + r) * K + c8);
  }
  wait_async0();
  __syncthreads();

  // ---- per-wave B pointers for NT column tiles ----
  const __bf16* bp[NT];
#pragma unroll
  for (int t = 0; t < NT; ++t)
    bp[t] = Bt + (size_t)(tn + t * 16 + l15) * K + (hi << 3);

  v8f acc[NT];
#pragma unroll
  for (int t = 0; t < NT; ++t) acc[t] = (v8f)0.0f;

  const __bf16* a_lds = &sA[l15 * KP + (hi << 3)];
#pragma unroll 2
  for (int k = 0; k < K; k += 32) {
    v8bf a0 = *(const v8bf*)(a_lds);        // K = k+hi*8 .. +7
    v8bf a1 = *(const v8bf*)(a_lds + 16);   // K = k+16+hi*8 .. +7
    v16bf af = __builtin_shufflevector(a0, a1, 0,1,2,3,4,5,6,7,8,9,10,11,12,13,14,15);
#pragma unroll
    for (int t = 0; t < NT; ++t) {
      v8bf b0 = gload((const v8bf*)(bp[t]));       // global_load_b128
      v8bf b1 = gload((const v8bf*)(bp[t]) + 2);   // global_load_b128
      gprefetch(bp[t] + 32);                       // global_prefetch_b8
      v16bf bv = __builtin_shufflevector(b0, b1, 0,1,2,3,4,5,6,7,8,9,10,11,12,13,14,15);
      acc[t] = __builtin_amdgcn_wmma_f32_16x16x32_bf16(
          /*neg_a=*/false, af, /*neg_b=*/false, bv,
          /*c_mod=*/(short)0, acc[t], /*reuse_a=*/false, /*reuse_b=*/false);
      bp[t] += 32;
    }
    a_lds += 32;
  }

#pragma unroll
  for (int t = 0; t < NT; ++t) {
    const int cb = tn + t * 16 + l15;
    const float bv = bias ? bias[cb] : 0.0f;
    float* cp = C + (size_t)(tm + (hi << 3)) * Nout + cb;
#pragma unroll
    for (int r = 0; r < 8; ++r) cp[(size_t)r * Nout] = acc[t][r] + bv;
  }
}

// ---------------- edge logits + segment max ----------------
// one block per edge; e_feat = edge_attr @ We computed inline from the
// transposed Wet[HC][16] (contiguous 64B per channel).
__global__ void __launch_bounds__(256)
k_edge_logits(const float* __restrict__ xl, const float* __restrict__ xr,
              const float* __restrict__ ea, const float* __restrict__ Wet,
              const float* __restrict__ att,
              const int* __restrict__ src, const int* __restrict__ dst,
              float* __restrict__ logits, float* __restrict__ segmax,
              int HC, int C) {
  const int e = blockIdx.x;
  const int s = src[e];
  const int d = dst[e];
  __shared__ float s_ea[EDIM];
  __shared__ float s_acc[NHEADS];
  if (threadIdx.x < EDIM)   s_ea[threadIdx.x]  = ea[(size_t)e * EDIM + threadIdx.x];
  if (threadIdx.x < NHEADS) s_acc[threadIdx.x] = 0.0f;
  __syncthreads();
  for (int j = threadIdx.x; j < HC; j += blockDim.x) {
    const v4f* wr = (const v4f*)(Wet + (size_t)j * EDIM);
    float ef = 0.0f;
#pragma unroll
    for (int q = 0; q < 4; ++q) {
      v4f w = gload(wr + q);
      ef += w.x * s_ea[4 * q] + w.y * s_ea[4 * q + 1] +
            w.z * s_ea[4 * q + 2] + w.w * s_ea[4 * q + 3];
    }
    float sv = gload(xl + (size_t)s * HC + j) + gload(xr + (size_t)d * HC + j) + ef;
    sv = sv >= 0.0f ? sv : 0.2f * sv;                 // leaky_relu(0.2)
    atomicAdd(&s_acc[j / C], sv * gload(att + j));    // ds_add_f32
  }
  __syncthreads();
  if (threadIdx.x < NHEADS) {
    float lg = s_acc[threadIdx.x];
    logits[(size_t)e * NHEADS + threadIdx.x] = lg;
    atomicMaxF(&segmax[(size_t)d * NHEADS + threadIdx.x], lg);
  }
}

__global__ void k_segmax_fix(float* __restrict__ m, long n) {
  long i = (long)blockIdx.x * blockDim.x + threadIdx.x;
  if (i < n && m[i] < -1e30f) m[i] = 0.0f;
}

// p = exp(logit - m[dst]); denom[dst] += p  (p overwrites logits buffer)
__global__ void k_edge_expsum(const int* __restrict__ dst, const float* __restrict__ segmax,
                              float* __restrict__ logits, float* __restrict__ denom, long EH) {
  long i = (long)blockIdx.x * blockDim.x + threadIdx.x;
  if (i >= EH) return;
  int e = (int)(i >> 2), h = (int)(i & 3);
  int d = dst[e];
  float p = __expf(logits[i] - gload(segmax + (size_t)d * NHEADS + h));
  logits[i] = p;
  atomicAdd(&denom[(size_t)d * NHEADS + h], p);
}

// out[dst] += x_l[src] * alpha ; one block per edge
__global__ void __launch_bounds__(256)
k_edge_scatter(const float* __restrict__ xl, const float* __restrict__ p,
               const float* __restrict__ denom,
               const int* __restrict__ src, const int* __restrict__ dst,
               float* __restrict__ out, int HC, int C) {
  const int e = blockIdx.x;
  const int s = src[e];
  const int d = dst[e];
  __shared__ float s_alpha[NHEADS];
  if (threadIdx.x < NHEADS)
    s_alpha[threadIdx.x] = gload(p + (size_t)e * NHEADS + threadIdx.x) /
                           (gload(denom + (size_t)d * NHEADS + threadIdx.x) + 1e-16f);
  __syncthreads();
  for (int j = threadIdx.x; j < HC; j += blockDim.x)
    atomicAdd(&out[(size_t)d * HC + j], gload(xl + (size_t)s * HC + j) * s_alpha[j / C]);
}

// ---------------- pooling + head ----------------
__global__ void __launch_bounds__(256)
k_pool(const float* __restrict__ h, const int* __restrict__ batch,
       float* __restrict__ xadd, float* __restrict__ cnt) {
  int n = blockIdx.x;
  int b = batch[n];
  atomicAdd(&xadd[(size_t)b * HC2 + threadIdx.x], gload(h + (size_t)n * HC2 + threadIdx.x));
  if (threadIdx.x == 0) atomicAdd(&cnt[b], 1.0f);
}

__global__ void __launch_bounds__(64)
k_head(const float* __restrict__ xadd, const float* __restrict__ cnt,
       const float* __restrict__ fc1w, const float* __restrict__ fc1b,
       const float* __restrict__ fc2w, const float* __restrict__ fc2b,
       float* __restrict__ out) {
  int g = blockIdx.x, t = threadIdx.x;
  __shared__ float s_h[64];
  float inv = 1.0f / fmaxf(cnt[g], 1.0f);
  float acc = fc1b[t];
  for (int k = 0; k < HC2; ++k) {
    float zs = xadd[(size_t)g * HC2 + k];
    acc += zs * fc1w[(size_t)k * 64 + t] + (zs * inv) * fc1w[(size_t)(HC2 + k) * 64 + t];
  }
  s_h[t] = acc > 0.0f ? acc : 0.0f;
  __syncthreads();
  if (t == 0) {
    float o = fc2b[0];
    for (int k = 0; k < 64; ++k) o += s_h[k] * fc2w[k];
    out[g] = o;
  }
}

// ---------------- launch ----------------
static inline long cdiv(long a, long b) { return (a + b - 1) / b; }

extern "C" void kernel_launch(void* const* d_in, const int* in_sizes, int n_in,
                              void* d_out, int out_size, void* d_ws, size_t ws_size,
                              hipStream_t stream) {
  (void)in_sizes; (void)n_in; (void)out_size; (void)ws_size;
  const float* x     = (const float*)d_in[0];
  const float* eattr = (const float*)d_in[1];
  const int*   src   = (const int*)d_in[2];              // edge_index[0]
  const int*   dst   = ((const int*)d_in[2]) + NEDGES;   // edge_index[1]
  const int*   batch = (const int*)d_in[3];
  const float* W1l = (const float*)d_in[4],  *b1l = (const float*)d_in[5];
  const float* W1r = (const float*)d_in[6],  *b1r = (const float*)d_in[7];
  const float* W1e = (const float*)d_in[8],  *att1 = (const float*)d_in[9];
  const float* bias1 = (const float*)d_in[10];
  const float* W2l = (const float*)d_in[11], *b2l = (const float*)d_in[12];
  const float* W2r = (const float*)d_in[13], *b2r = (const float*)d_in[14];
  const float* W2e = (const float*)d_in[15], *att2 = (const float*)d_in[16];
  const float* bias2 = (const float*)d_in[17];
  const float* fc1w = (const float*)d_in[18], *fc1b = (const float*)d_in[19];
  const float* fc2w = (const float*)d_in[20], *fc2b = (const float*)d_in[21];
  float* out = (float*)d_out;

  // ---- workspace carve ----
  char* ws = (char*)d_ws;
  size_t o = 0;
  auto carve = [&](size_t bytes) -> void* {
    void* p = ws + o;
    o = (o + bytes + 255) & ~(size_t)255;
    return p;
  };
  __bf16* xb   = (__bf16*)carve((size_t)NNODES * 64 * 2);
  __bf16* W1lt = (__bf16*)carve((size_t)HC1 * 64 * 2);
  __bf16* W1rt = (__bf16*)carve((size_t)HC1 * 64 * 2);
  __bf16* W2lt = (__bf16*)carve((size_t)HC2 * HC1 * 2);
  __bf16* W2rt = (__bf16*)carve((size_t)HC2 * HC1 * 2);
  float*  W1et = (float*)carve((size_t)HC1 * EDIM * 4);
  float*  W2et = (float*)carve((size_t)HC2 * EDIM * 4);
  float*  xl1  = (float*)carve((size_t)NNODES * HC1 * 4);
  float*  xr1  = (float*)carve((size_t)NNODES * HC1 * 4);
  float*  lg1  = (float*)carve((size_t)NEDGES * NHEADS * 4);
  float*  sm1  = (float*)carve((size_t)NNODES * NHEADS * 4);
  float*  dn1  = (float*)carve((size_t)NNODES * NHEADS * 4);
  float*  h1   = (float*)carve((size_t)NNODES * HC1 * 4);
  __bf16* h1b  = (__bf16*)carve((size_t)NNODES * HC1 * 2);
  float*  xl2  = (float*)carve((size_t)NNODES * HC2 * 4);
  float*  xr2  = (float*)carve((size_t)NNODES * HC2 * 4);
  float*  lg2  = (float*)carve((size_t)NEDGES * NHEADS * 4);
  float*  sm2  = (float*)carve((size_t)NNODES * NHEADS * 4);
  float*  dn2  = (float*)carve((size_t)NNODES * NHEADS * 4);
  float*  h2   = (float*)carve((size_t)NNODES * HC2 * 4);
  float*  xadd = (float*)carve((size_t)NGRAPH * HC2 * 4);
  float*  cnt  = (float*)carve((size_t)NGRAPH * 4);

  const int T = 256;
  // ---- prep: casts / transposes ----
  k_cast_bf16<<<cdiv((long)NNODES * 64, T), T, 0, stream>>>(x, xb, (long)NNODES * 64);
  k_transpose_cast<<<cdiv((long)64 * HC1, T), T, 0, stream>>>(W1l, W1lt, 64, HC1);
  k_transpose_cast<<<cdiv((long)64 * HC1, T), T, 0, stream>>>(W1r, W1rt, 64, HC1);
  k_transpose_cast<<<cdiv((long)HC1 * HC2, T), T, 0, stream>>>(W2l, W2lt, HC1, HC2);
  k_transpose_cast<<<cdiv((long)HC1 * HC2, T), T, 0, stream>>>(W2r, W2rt, HC1, HC2);
  k_transpose_f32<<<cdiv((long)EDIM * HC1, T), T, 0, stream>>>(W1e, W1et, EDIM, HC1);
  k_transpose_f32<<<cdiv((long)EDIM * HC2, T), T, 0, stream>>>(W2e, W2et, EDIM, HC2);

  // ---- layer 1: dense transforms (WMMA, NT=4 -> block covers 16x512) ----
  dim3 g1(NNODES / 16, HC1 / 512);
  k_gemm_wmma_bf16<4><<<g1, T, 0, stream>>>(xb, W1lt, b1l, xl1, NNODES, 64, HC1);
  k_gemm_wmma_bf16<4><<<g1, T, 0, stream>>>(xb, W1rt, b1r, xr1, NNODES, 64, HC1);

  // ---- layer 1: edge softmax + aggregation ----
  long nh = (long)NNODES * NHEADS;
  k_fill<<<cdiv(nh, T), T, 0, stream>>>(sm1, -3e38f, nh);
  k_fill<<<cdiv(nh, T), T, 0, stream>>>(dn1, 0.0f, nh);
  k_edge_logits<<<NEDGES, T, 0, stream>>>(xl1, xr1, eattr, W1et, att1, src, dst, lg1, sm1, HC1, CC1);
  k_segmax_fix<<<cdiv(nh, T), T, 0, stream>>>(sm1, nh);
  k_edge_expsum<<<cdiv((long)NEDGES * NHEADS, T), T, 0, stream>>>(dst, sm1, lg1, dn1, (long)NEDGES * NHEADS);
  k_bias_init<<<cdiv((long)NNODES * HC1, T), T, 0, stream>>>(bias1, h1, HC1, (long)NNODES * HC1);
  k_edge_scatter<<<NEDGES, T, 0, stream>>>(xl1, lg1, dn1, src, dst, h1, HC1, CC1);
  k_relu_cast<<<cdiv((long)NNODES * HC1, T), T, 0, stream>>>(h1, h1b, (long)NNODES * HC1);

  // ---- layer 2: dense transforms (WMMA, NT=2 -> block covers 16x256) ----
  dim3 g2(NNODES / 16, HC2 / 256);
  k_gemm_wmma_bf16<2><<<g2, T, 0, stream>>>(h1b, W2lt, b2l, xl2, NNODES, HC1, HC2);
  k_gemm_wmma_bf16<2><<<g2, T, 0, stream>>>(h1b, W2rt, b2r, xr2, NNODES, HC1, HC2);

  // ---- layer 2: edge softmax + aggregation ----
  k_fill<<<cdiv(nh, T), T, 0, stream>>>(sm2, -3e38f, nh);
  k_fill<<<cdiv(nh, T), T, 0, stream>>>(dn2, 0.0f, nh);
  k_edge_logits<<<NEDGES, T, 0, stream>>>(xl2, xr2, eattr, W2et, att2, src, dst, lg2, sm2, HC2, CC2);
  k_segmax_fix<<<cdiv(nh, T), T, 0, stream>>>(sm2, nh);
  k_edge_expsum<<<cdiv((long)NEDGES * NHEADS, T), T, 0, stream>>>(dst, sm2, lg2, dn2, (long)NEDGES * NHEADS);
  k_bias_init<<<cdiv((long)NNODES * HC2, T), T, 0, stream>>>(bias2, h2, HC2, (long)NNODES * HC2);
  k_edge_scatter<<<NEDGES, T, 0, stream>>>(xl2, lg2, dn2, src, dst, h2, HC2, CC2);
  k_relu_cast<<<cdiv((long)NNODES * HC2, T), T, 0, stream>>>(h2, (__bf16*)nullptr, (long)NNODES * HC2);

  // ---- pooling + head ----
  k_fill<<<cdiv((long)NGRAPH * HC2, T), T, 0, stream>>>(xadd, 0.0f, (long)NGRAPH * HC2);
  k_fill<<<1, 64, 0, stream>>>(cnt, 0.0f, NGRAPH);
  k_pool<<<NNODES, HC2, 0, stream>>>(h2, batch, xadd, cnt);
  k_head<<<NGRAPH, 64, 0, stream>>>(xadd, cnt, fc1w, fc1b, fc2w, fc2b, out);
}
